// ATACGCN_83820581749460
// MI455X (gfx1250) — matrix operand order repaired and checked by reading, not compiled, not run on hardware
//
#include <hip/hip_runtime.h>
#include <hip/hip_bf16.h>

// ---------------------------------------------------------------------------
// ATACGCN on MI455X (gfx1250): 2-layer GCN, adj(16384x16384,f32) dominant.
//  - s/u streams fused -> one (16384 x 128) right operand: adj streamed from
//    HBM exactly once per layer (2 x 1GiB ~= 92us @ 23.3TB/s = the roofline).
//  - bf16 WMMA (v_wmma_f32_16x16x32_bf16), f32 accumulate; adj converted
//    f32->bf16 in-register during the LDS tile fill.
//  - XW kept TRANSPOSED in the workspace (XWT[col][row]) so the spmm B-tile
//    fill is vectorized b128 traffic instead of scalar ds_store_b16 scatter.
//  - spmm uses register-staged double buffering: next tile's global loads
//    issue before this tile's WMMAs; one barrier per K-step.
// ---------------------------------------------------------------------------

constexpr int kN   = 16384;   // nodes
constexpr int kD   = 64;      // feature dim per stream
constexpr int kC   = 128;     // packed columns (s|u)
constexpr int kLDA = 80;      // padded LDS K-stride (bf16): 160B, 32B-aligned
constexpr int kLDT = 136;     // padded LDS row-stride for xw transpose staging

typedef __bf16 bf16_t;
typedef __attribute__((ext_vector_type(16))) __bf16 v16bf;
typedef __attribute__((ext_vector_type(8)))  __bf16 v8bf;
typedef __attribute__((ext_vector_type(4)))  __bf16 v4bf;
typedef __attribute__((ext_vector_type(8)))  float  v8f;
typedef __attribute__((ext_vector_type(4)))  float  v4f;

__device__ __forceinline__ float elu_f(float x) {
    return x > 0.0f ? x : (__expf(x) - 1.0f);
}

__device__ __forceinline__ v8f wmma_bf16(v16bf a, v16bf b, v8f c) {
    // (neg_a, A, neg_b, B, c_mod, C, reuse_a, reuse_b)
    return __builtin_amdgcn_wmma_f32_16x16x32_bf16(false, a, false, b,
                                                   (short)0, c, false, false);
}

union FragA { v16bf v; struct { v8bf lo, hi; } h; };

// ---------------------------------------------------------------------------
// Kernel 1: X0[n, c] = bf16(elu(z[n, c])) for c in [0,128)  (zs | zu packed)
// ---------------------------------------------------------------------------
__global__ __launch_bounds__(256) void prep_kernel(const float* __restrict__ z,
                                                   bf16_t* __restrict__ X0) {
    int idx = blockIdx.x * 256 + threadIdx.x;        // < kN*kC
    int n = idx >> 7, c = idx & 127;
    float v = z[(size_t)n * 192 + c];
    X0[idx] = (bf16_t)elu_f(v);
}

// ---------------------------------------------------------------------------
// Kernel 2: XWT[h*64+c][:] = (X[:, h*64 : h*64+64] @ W[:, c])^T   (h = 0,1)
// WG = 128 rows x 128 cols, 8 waves; wave w owns M-subtile w, all 8 N-subtiles.
// W (64x64 f32) -> bf16, transposed in LDS (lW[c][k]).  C is staged through
// LDS (sT[col][row]) so the column-major global stores are coalesced b128s.
// ---------------------------------------------------------------------------
__global__ __launch_bounds__(256) void xw_kernel(const bf16_t* __restrict__ X,
                                                 const float* __restrict__ W,
                                                 bf16_t* __restrict__ XWT) {
    __shared__ __attribute__((aligned(32))) bf16_t lW[kD][kLDA];
    __shared__ __attribute__((aligned(32))) bf16_t sT[kC][kLDT];  // [col][row]

    const int t    = threadIdx.x;
    const int wave = t >> 5, lane = t & 31;
    const int m    = lane & 15, kg = lane >> 4;
    const int rowBase = blockIdx.x * 128;

    #pragma unroll
    for (int i = 0; i < 16; ++i) {                   // 4096 elems / 256 thr
        int e = t + 256 * i;
        int k = e >> 6, c = e & 63;
        lW[c][k] = (bf16_t)W[e];                     // W row-major [k][c] -> lW[c][k]
    }
    __syncthreads();

    v8f acc[8] = {};
    const bf16_t* rp = X + (size_t)(rowBase + wave * 16 + m) * kC;

    #pragma unroll
    for (int ks = 0; ks < 2; ++ks) {                 // K = 64 -> 2 WMMA K-steps
        FragA a0, a1;
        a0.h.lo = *(const v8bf*)&rp[      ks * 32 + kg * 16];
        a0.h.hi = *(const v8bf*)&rp[      ks * 32 + kg * 16 + 8];
        a1.h.lo = *(const v8bf*)&rp[64 +  ks * 32 + kg * 16];
        a1.h.hi = *(const v8bf*)&rp[64 +  ks * 32 + kg * 16 + 8];
        #pragma unroll
        for (int j = 0; j < 4; ++j) {
            v16bf b = *(const v16bf*)&lW[j * 16 + m][ks * 32 + kg * 16];
            acc[j]     = wmma_bf16(a0.v, b, acc[j]);      // half 0 (s-stream)
            acc[j + 4] = wmma_bf16(a1.v, b, acc[j + 4]);  // half 1 (u-stream)
        }
    }

    // stage C transposed in LDS, then coalesced column-major global stores
    #pragma unroll
    for (int j = 0; j < 8; ++j) {
        int col = (j >> 2) * 64 + (j & 3) * 16 + m;
        #pragma unroll
        for (int r = 0; r < 8; ++r)
            sT[col][wave * 16 + kg * 8 + r] = (bf16_t)acc[j][r];
    }
    __syncthreads();
    #pragma unroll
    for (int i = 0; i < 8; ++i) {                    // 2048 b128 chunks / 256 thr
        int e = t + 256 * i;
        int col = e >> 4, r8 = (e & 15) << 3;
        *(v8bf*)&XWT[(size_t)col * kN + rowBase + r8] = *(const v8bf*)&sT[col][r8];
    }
}

// ---------------------------------------------------------------------------
// Kernel 3: Y = bf16(elu(adj @ XW + bias))          [the 34-GFLOP pass, x2]
// WG tile = 64 rows x 128 cols; K blocked by 64; 256 WGs -> adj streamed once.
// Register-staged double buffering: loads for tile k+1 issue before the
// WMMAs on tile k; LDS stores for k+1 land after; one barrier per K-step.
// ---------------------------------------------------------------------------
__global__ __launch_bounds__(256) void spmm_elu_kernel(const float* __restrict__ adj,
                                                       const bf16_t* __restrict__ XWT,
                                                       const float* __restrict__ bias,
                                                       bf16_t* __restrict__ Y) {
    __shared__ __attribute__((aligned(32))) bf16_t lA[2][64][kLDA];   // [m][k]
    __shared__ __attribute__((aligned(32))) bf16_t lB[2][kC][kLDA];   // [n][k]

    const int t    = threadIdx.x;
    const int wave = t >> 5, lane = t & 31;
    const int ms   = wave & 3;            // M-subtile (16 rows) within 64
    const int nh   = wave >> 2;           // column half: N-subtiles nh*4 .. nh*4+3
    const int m    = lane & 15, kg = lane >> 4;
    const int rowBase = blockIdx.x * 64;

    // per-thread tile-fill coordinates (fixed across K-steps)
    int ar[4], ak[4], bn[4], bk[4];
    #pragma unroll
    for (int i = 0; i < 4; ++i) {
        int c = t + 256 * i;
        ar[i] = c >> 4;  ak[i] = (c & 15) << 2;   // adj: 64 rows x 64 k (f32)
        bn[i] = c >> 3;  bk[i] = (c & 7)  << 3;   // XWT: 128 n  x 64 k (bf16)
    }

    v4f  aReg[4];
    v8bf bReg[4];

    auto load_tiles = [&](int kb) {
        #pragma unroll
        for (int i = 0; i < 4; ++i)
            aReg[i] = *(const v4f*)&adj[(size_t)(rowBase + ar[i]) * kN + kb + ak[i]];
        #pragma unroll
        for (int i = 0; i < 4; ++i)
            bReg[i] = *(const v8bf*)&XWT[(size_t)bn[i] * kN + kb + bk[i]];
    };
    auto stage_tiles = [&](int buf) {
        #pragma unroll
        for (int i = 0; i < 4; ++i) {
            v4bf h;
            h[0] = (bf16_t)aReg[i][0]; h[1] = (bf16_t)aReg[i][1];
            h[2] = (bf16_t)aReg[i][2]; h[3] = (bf16_t)aReg[i][3];
            *(v4bf*)&lA[buf][ar[i]][ak[i]] = h;
        }
        #pragma unroll
        for (int i = 0; i < 4; ++i)
            *(v8bf*)&lB[buf][bn[i]][bk[i]] = bReg[i];
    };

    v8f acc[4] = {};

    load_tiles(0);
    stage_tiles(0);
    int cur = 0;
    for (int kb = 0; kb < kN; kb += 64) {
        __syncthreads();                       // buf[cur] visible to all waves
        const bool more = (kb + 64) < kN;
        if (more) load_tiles(kb + 64);         // issue next global loads early
        #pragma unroll
        for (int ks = 0; ks < 2; ++ks) {
            FragA a;   // A 16x32 layout: lane m holds 16 contiguous K starting at kg*16
            a.h.lo = *(const v8bf*)&lA[cur][ms * 16 + m][ks * 32 + kg * 16];
            a.h.hi = *(const v8bf*)&lA[cur][ms * 16 + m][ks * 32 + kg * 16 + 8];
            #pragma unroll
            for (int j = 0; j < 4; ++j) {
                // B 32x16 layout: lane n holds 16 contiguous K starting at kg*16
                v16bf b = *(const v16bf*)&lB[cur][(nh * 4 + j) * 16 + m][ks * 32 + kg * 16];
                acc[j] = wmma_bf16(a.v, b, acc[j]);
            }
        }
        if (more) stage_tiles(cur ^ 1);        // stage next tile in other buffer
        cur ^= 1;
    }

    // epilogue: + bias, elu, store bf16.  C layout: row = r + 8*kg, col = lane&15.
    #pragma unroll
    for (int j = 0; j < 4; ++j) {
        int col = (nh * 4 + j) * 16 + m;
        float bv = bias[col & 63];
        #pragma unroll
        for (int r = 0; r < 8; ++r) {
            int row = rowBase + ms * 16 + kg * 8 + r;
            float v = acc[j][r] + bv;
            Y[(size_t)row * kC + col] = (bf16_t)elu_f(v);
        }
    }
}

// ---------------------------------------------------------------------------
// Kernel 4: out[n,o] = ys@Wl[0:64] + yu@Wl[64:128] + za@Wl[128:192] + bl
// Wl (192x64 f32 = 48KB) staged in LDS; 4 rows x 64 cols per 256-thread block.
// ---------------------------------------------------------------------------
__global__ __launch_bounds__(256) void final_kernel(const bf16_t* __restrict__ Y,
                                                    const float* __restrict__ z,
                                                    const float* __restrict__ Wl,
                                                    const float* __restrict__ bl,
                                                    float* __restrict__ out) {
    __shared__ float sW[192 * 64];
    __shared__ float sb[64];
    const int t = threadIdx.x;
    for (int e = t; e < 192 * 64; e += 256) sW[e] = Wl[e];
    if (t < 64) sb[t] = bl[t];
    __syncthreads();

    const int o = t & 63, rl = t >> 6;
    const int row = blockIdx.x * 4 + rl;
    const bf16_t* yr = Y + (size_t)row * kC;
    const float*  zr = z + (size_t)row * 192 + 128;

    float acc = sb[o];
    #pragma unroll 8
    for (int k = 0; k < 64; ++k) acc += (float)yr[k]      * sW[k * 64 + o];
    #pragma unroll 8
    for (int k = 0; k < 64; ++k) acc += (float)yr[64 + k] * sW[(64 + k) * 64 + o];
    #pragma unroll 8
    for (int k = 0; k < 64; ++k) acc += zr[k]             * sW[(128 + k) * 64 + o];
    out[(size_t)row * 64 + o] = acc;
}

// ---------------------------------------------------------------------------
extern "C" void kernel_launch(void* const* d_in, const int* in_sizes, int n_in,
                              void* d_out, int out_size, void* d_ws, size_t ws_size,
                              hipStream_t stream) {
    (void)in_sizes; (void)n_in; (void)out_size; (void)ws_size;
    const float* z   = (const float*)d_in[0];   // (N, 192)
    const float* adj = (const float*)d_in[1];   // (N, N)
    const float* Ws  = (const float*)d_in[2];   // (2, 64, 64)
    const float* bs  = (const float*)d_in[3];   // (2, 64)
    const float* Wl  = (const float*)d_in[4];   // (192, 64)
    const float* bl  = (const float*)d_in[5];   // (64,)
    float* out = (float*)d_out;

    // workspace: three N x 128 bf16 buffers (4 MB each, 12 MB total)
    bf16_t* X0  = (bf16_t*)d_ws;
    bf16_t* X1  = X0 + (size_t)kN * kC;
    bf16_t* XWT = X1 + (size_t)kN * kC;        // transposed: [col][row]

    prep_kernel<<<(kN * kC) / 256, 256, 0, stream>>>(z, X0);

    // layer 0:  X1 = elu(adj @ (X0 @ Ws[0]) + bs[0])
    xw_kernel<<<kN / 128, 256, 0, stream>>>(X0, Ws, XWT);
    spmm_elu_kernel<<<kN / 64, 256, 0, stream>>>(adj, XWT, bs, X1);

    // layer 1:  X0 = elu(adj @ (X1 @ Ws[1]) + bs[1])
    xw_kernel<<<kN / 128, 256, 0, stream>>>(X1, Ws + 64 * 64, XWT);
    spmm_elu_kernel<<<kN / 64, 256, 0, stream>>>(adj, XWT, bs + 64, X0);

    // head: out = [ys | yu | za] @ Wl + bl
    final_kernel<<<kN / 4, 256, 0, stream>>>(X0, z, Wl, bl, out);
}
